// ZBLRepulsion_3616362463303
// MI455X (gfx1250) — matrix operand order repaired and checked by reading, not compiled, not run on hardware
//
#include <hip/hip_runtime.h>

typedef __attribute__((ext_vector_type(16))) _Float16 v16h;
typedef __attribute__((ext_vector_type(8)))  float    v8f;

// ---- physical constants from the reference ----
#define KE_CONST   14.3996454784255f
#define INV_A_PRE  ((float)(1.0 / (0.88534 * 0.52917721092)))   // 1/(0.88534*a0)
#define ZA0 0.1818f
#define ZA1 0.5099f
#define ZA2 0.2802f
#define ZA3 0.02817f
#define ZB0 3.2f
#define ZB1 0.9423f
#define ZB2 0.4029f
#define ZB3 0.2016f

#define WAVES_PER_BLOCK 8
#define MAX_ELEM 96            // padded LDS row stride (>= n_elem, 3*32)

// ---------------- async global->LDS copy (CDNA5 GLOBAL_LOAD_ASYNC_TO_LDS) ----
#if __has_builtin(__builtin_amdgcn_global_load_async_to_lds_b32)
#define HAVE_ASYNC_LDS 1
#else
#define HAVE_ASYNC_LDS 0
#endif

typedef __attribute__((address_space(1))) int GI;   // global int
typedef __attribute__((address_space(3))) int LI;   // LDS int

__device__ __forceinline__ void tile_copy_b32(const float* g, float* l)
{
#if HAVE_ASYNC_LDS
    __builtin_amdgcn_global_load_async_to_lds_b32((GI*)g, (LI*)l, 0, 0);
#else
    *l = *g;
#endif
}

__device__ __forceinline__ void tile_copy_wait()
{
#if HAVE_ASYNC_LDS
#if __has_builtin(__builtin_amdgcn_s_wait_asynccnt)
    __builtin_amdgcn_s_wait_asynccnt(0);
#else
    asm volatile("s_wait_asynccnt 0" ::: "memory");
#endif
#endif
}

// ------------------------------------------------------------------
// Kernel 0: zero the output accumulator (harness poisons d_out).
// ------------------------------------------------------------------
__global__ __launch_bounds__(256) void zbl_zero_kernel(float* __restrict__ out, int n)
{
    int i = blockIdx.x * 256 + threadIdx.x;
    if (i < n) out[i] = 0.0f;
}

// ------------------------------------------------------------------
// Kernel 1: Z_node = node_attrs @ atomic_numbers via v_wmma_f32_16x16x32_f16.
// One wave = one 16-node tile. LDS tile is row-stride-96 and zero-padded,
// so the WMMA operand pack is pure constant-offset ds loads (no selects).
// One-hot (0/1) and Z (1..89) are exact in f16 -> exact f32 result.
// Stores float2 {Z, Z^0.23} per node into d_ws.
// ------------------------------------------------------------------
__global__ __launch_bounds__(256) void zbl_ztab_kernel(
    const float* __restrict__ node_attrs,      // [n_nodes * n_elem]
    const float* __restrict__ atomic_numbers,  // [n_elem]
    float2*      __restrict__ ztab,            // [n_nodes] {Z, Z^0.23}
    int n_nodes, int n_elem)
{
    __shared__ float s_tile[WAVES_PER_BLOCK][16 * MAX_ELEM];  // padded tiles
    __shared__ float s_bvec[MAX_ELEM];                        // padded Z vector

    const int tid       = threadIdx.x;
    const int waveInBlk = tid >> 5;
    const int lane      = tid & 31;
    const int tile      = blockIdx.x * WAVES_PER_BLOCK + waveInBlk;
    const bool hi       = (lane >= 16);

    float* wtile = s_tile[waveInBlk];

    // ---- stage padded atomic_numbers once per block (zero-filled tail) ----
    if (tid < MAX_ELEM) {
        float v = 0.0f;
        if (tid < n_elem) v = atomic_numbers[tid];
        s_bvec[tid] = v;
    }

    // ---- zero the pad cells (cols in [MAX_ELEM-8, MAX_ELEM) that are >= n_elem).
    //      These cells are never written by the copy below -> no ordering hazard.
    #pragma unroll
    for (int p = 0; p < 4; ++p) {                  // 4*32 = 128 = 16 rows * 8 cols
        const int cell = p * 32 + lane;
        const int row  = cell >> 3;
        const int col  = (MAX_ELEM - 8) + (cell & 7);
        if (col >= n_elem) wtile[row * MAX_ELEM + col] = 0.0f;
    }

    // ---- async coalesced copy: 16 rows, cols {lane, lane+32, lane+64} ----
    const long long lastRow = (long long)n_nodes - 1;
    for (int roff = 0; roff < 16; ++roff) {
        long long grow = (long long)tile * 16 + roff;
        grow = (grow > lastRow) ? lastRow : grow;           // clamp tail waves
        const float* gsrc = node_attrs + grow * n_elem;
        float*       ldst = wtile + roff * MAX_ELEM;
        #pragma unroll
        for (int cc = 0; cc < 3; ++cc) {
            const int col = lane + cc * 32;
            if (col < n_elem)                               // EXEC-masked async op
                tile_copy_b32(gsrc + col, ldst + col);
        }
    }
    tile_copy_wait();
    __syncthreads();

    // ---- pack + WMMA: all LDS reads are unconditional, constant offsets ----
    const float* rowl = wtile + (lane & 15) * MAX_ELEM;     // 16B-aligned base

    v8f acc = {};
    #pragma unroll
    for (int c = 0; c < 3; ++c) {
        // A operand, ISA 16-bit 16x32 layout:
        //  lanes 0-15 : halves 0..7 = K kb..kb+7, halves 8..15 = K kb+16..kb+23 (kb=32c)
        //  lanes 16-31: same with kb = 32c + 8
        const int kb = c * 32 + (hi ? 8 : 0);
        v16h a;
        #pragma unroll
        for (int j = 0; j < 8; ++j) {
            a[j]     = (_Float16)rowl[kb + j];
            a[8 + j] = (_Float16)rowl[kb + 16 + j];
        }
        // B operand, ISA 16-bit 32x16 layout, broadcast to all 16 columns:
        //  same K order as A: halves 0..7 = K bb..bb+7, halves 8..15 = K bb+16..bb+23; bb = 32c (+8 for hi lanes)
        const int bb = c * 32 + (hi ? 8 : 0);
        v16h b;
        #pragma unroll
        for (int j = 0; j < 16; ++j) {
            b[j] = (_Float16)s_bvec[bb + (j < 8 ? j : 8 + j)];
        }
        acc = __builtin_amdgcn_wmma_f32_16x16x32_f16(
            /*neg_a=*/false, a, /*neg_b=*/false, b,
            /*c_mod=*/(short)0, acc, /*reuse_a=*/false, /*reuse_b=*/false);
    }

    // C/D layout: VGPR v, lane L -> row M = v + 8*(L>=16), col N = L&15.
    // Broadcast B => all columns identical. Lanes {0..7, 16..23} emit one row each.
    if ((lane & 8) == 0) {
        const int v = lane & 7;
        float z = acc[0];
        #pragma unroll
        for (int i = 1; i < 8; ++i) if (v == i) z = acc[i];
        const int outRow = tile * 16 + v + (hi ? 8 : 0);
        if (outRow < n_nodes) {
            float2 zz;
            zz.x = z;
            zz.y = __powf(z, 0.23f);
            ztab[outRow] = zz;
        }
    }
}

// ------------------------------------------------------------------
// Kernel 2: per-edge ZBL energy, scatter-add halves to both endpoints.
// Memory-bound pass (roofline-dominant): 4B length + 16B int64 indices
// streamed per edge (~260MB total -> ~11us floor at 23.3 TB/s);
// {Z, Z^0.23} gathers hit the 1.6MB L2-resident table; 2 f32 atomics/edge.
// ------------------------------------------------------------------
__global__ __launch_bounds__(256) void zbl_edge_kernel(
    const float*      __restrict__ lengths,     // [E]
    const long long*  __restrict__ edge_index,  // [2*E] (src row, then dst row)
    const float2*     __restrict__ ztab,        // [N] {Z, Z^0.23}
    const float*      __restrict__ r_max,       // [1]
    float*            __restrict__ out,         // [N]
    long long n_edges)
{
    const long long e = (long long)blockIdx.x * 256 + threadIdx.x;
    if (e >= n_edges) return;

    const float r  = fmaxf(lengths[e], 0.2f);
    const int   s  = (int)edge_index[e];
    const int   d  = (int)edge_index[n_edges + e];
    const float2 zi = ztab[s];
    const float2 zj = ztab[d];

    // x = r / a,  a = 0.88534*a0 / (Zi^0.23 + Zj^0.23)
    const float x = r * (zi.y + zj.y) * INV_A_PRE;

    const float phi = ZA0 * __expf(-ZB0 * x)
                    + ZA1 * __expf(-ZB1 * x)
                    + ZA2 * __expf(-ZB2 * x)
                    + ZA3 * __expf(-ZB3 * x);

    const float V = KE_CONST * zi.x * zj.x * phi / r;

    // (1 - clip(r/rmax, 0, 1))^6
    const float t   = 1.0f - fminf(r / r_max[0], 1.0f);
    const float t2  = t * t;
    const float cut = t2 * t2 * t2;

    const float half = 0.25f * V * cut;   // 0.5 * 0.5 * V

    // fire-and-forget f32 atomics -> global_atomic_add_f32 (no return)
    unsafeAtomicAdd(out + s, half);
    unsafeAtomicAdd(out + d, half);
}

// ------------------------------------------------------------------
// Launcher
// ------------------------------------------------------------------
extern "C" void kernel_launch(void* const* d_in, const int* in_sizes, int n_in,
                              void* d_out, int out_size, void* d_ws, size_t ws_size,
                              hipStream_t stream)
{
    const float*     lengths        = (const float*)d_in[0];
    const float*     node_attrs     = (const float*)d_in[1];
    const float*     atomic_numbers = (const float*)d_in[2];
    const float*     r_max          = (const float*)d_in[3];
    const long long* edge_index     = (const long long*)d_in[4];

    const long long n_edges = (long long)in_sizes[0];
    const int       n_elem  = in_sizes[2];
    const int       n_nodes = in_sizes[1] / n_elem;

    float*  out  = (float*)d_out;
    float2* ztab = (float2*)d_ws;   // n_nodes * 8 bytes scratch

    // 1) zero accumulator
    {
        int blocks = (n_nodes + 255) / 256;
        zbl_zero_kernel<<<blocks, 256, 0, stream>>>(out, n_nodes);
    }
    // 2) WMMA matvec -> {Z, Z^0.23} table
    {
        int tiles  = (n_nodes + 15) / 16;                        // one wave per tile
        int blocks = (tiles + WAVES_PER_BLOCK - 1) / WAVES_PER_BLOCK;
        zbl_ztab_kernel<<<blocks, 256, 0, stream>>>(node_attrs, atomic_numbers,
                                                    ztab, n_nodes, n_elem);
    }
    // 3) edge pass with scatter atomics
    {
        long long blocks = (n_edges + 255) / 256;
        zbl_edge_kernel<<<(int)blocks, 256, 0, stream>>>(lengths, edge_index, ztab,
                                                         r_max, out, n_edges);
    }
}
